// GTLayer_8323646619704
// MI455X (gfx1250) — matrix operand order repaired and checked by reading, not compiled, not run on hardware
//
#include <hip/hip_runtime.h>

#define N_N 4096
#define DD  256
#define NH  8
#define HDD 32
#define EPSV 1e-5f

typedef __bf16 bf16_t;
typedef __attribute__((ext_vector_type(16))) __bf16 v16bf;
typedef __attribute__((ext_vector_type(8)))  float  v8f;

// ---------------------------------------------------------------------------
// WMMA core: D = A(16x32 bf16) x B(32x16 bf16) + C(16x16 f32)
// ---------------------------------------------------------------------------
__device__ inline v8f wmma_bf16(v16bf a, v16bf b, v8f c) {
  return __builtin_amdgcn_wmma_f32_16x16x32_bf16(false, a, false, b,
                                                 (short)0, c, false, false);
}

// A-matrix layout (16-bit, 16x32): lane L holds row (L&15), with
// K = (L>>4)*16 + 0..15 contiguous (matches the B layout).
__device__ inline v16bf load_a_bf16(const bf16_t* base, int stride, int lane) {
  int row = lane & 15, half = lane >> 4;
  const bf16_t* p = base + row * stride + half * 16;
  v16bf r;
  float4* r4 = reinterpret_cast<float4*>(&r);
  r4[0] = *reinterpret_cast<const float4*>(p);        // K = half*16 .. +7
  r4[1] = *reinterpret_cast<const float4*>(p + 8);    // K = half*16+8 .. +15
  return r;
}

__device__ inline v16bf load_a_f32(const float* base, int stride, int lane) {
  int row = lane & 15, half = lane >> 4;
  const float* p = base + row * stride + half * 16;
  v16bf r;
#pragma unroll
  for (int i = 0; i < 8; ++i) r[i]     = (bf16_t)p[i];
#pragma unroll
  for (int i = 0; i < 8; ++i) r[8 + i] = (bf16_t)p[8 + i];
  return r;
}

// B-matrix layout (16-bit, 32x16): lane L holds column n=(L&15), with
// K = (L>>4)*16 + 0..15 contiguous. Passing row-major rows gives B = rows^T.
__device__ inline v16bf load_b_rows_bf16(const bf16_t* rowbase, int stride, int lane) {
  int n = lane & 15, kb = (lane >> 4) << 4;
  const bf16_t* p = rowbase + n * stride + kb;
  v16bf r;
  float4* r4 = reinterpret_cast<float4*>(&r);
  r4[0] = *reinterpret_cast<const float4*>(p);
  r4[1] = *reinterpret_cast<const float4*>(p + 8);
  return r;
}

__device__ inline v16bf load_b_rows_f32(const float* rowbase, int stride, int lane) {
  int n = lane & 15, kb = (lane >> 4) << 4;
  const float* p = rowbase + n * stride + kb;
  v16bf r;
#pragma unroll
  for (int i = 0; i < 16; ++i) r[i] = (bf16_t)p[i];
  return r;
}

// GEMM inner loops over K (stride == row length for both operands here).
__device__ inline v8f gemm_f32ab(const float* Abase, const float* Brow,
                                 int stride, int ksteps, int lane) {
  v8f acc = {};
#pragma unroll
  for (int kk = 0; kk < ksteps; ++kk)
    acc = wmma_bf16(load_a_f32(Abase + kk * 32, stride, lane),
                    load_b_rows_f32(Brow + kk * 32, stride, lane), acc);
  return acc;
}

__device__ inline v8f gemm_bf16a(const bf16_t* Abase, const float* Brow,
                                 int stride, int ksteps, int lane) {
  v8f acc = {};
#pragma unroll
  for (int kk = 0; kk < ksteps; ++kk)
    acc = wmma_bf16(load_a_bf16(Abase + kk * 32, stride, lane),
                    load_b_rows_f32(Brow + kk * 32, stride, lane), acc);
  return acc;
}

// ---------------------------------------------------------------------------
// Kernel 0: zero BN accumulators (graph-replay determinism)
// ---------------------------------------------------------------------------
__global__ void k_zero(float* p, int n) {
  int i = blockIdx.x * blockDim.x + threadIdx.x;
  if (i < n) p[i] = 0.0f;
}

// ---------------------------------------------------------------------------
// Kernel 1: QKV projections. q pre-scaled by D^-0.5 = 1/16.
// Q,K stored per-head row-major [head][n][hd] (bf16). V stored transposed
// [head][hd][perm(n)] where within each 32-node chunk the node order is the
// interleaved K-permutation (n, n+16, n+1, n+17, ...) consumed by k_attn's
// packed-probs WMMA (any consistent K permutation of A and B is legal).
// Uniform (dst, base, stride-in-r) select -> branch-free store loop.
// ---------------------------------------------------------------------------
__global__ __launch_bounds__(256) void k_qkv(
    const float* __restrict__ h, const float* __restrict__ Wq,
    const float* __restrict__ Wk, const float* __restrict__ Wv,
    bf16_t* __restrict__ Qr, bf16_t* __restrict__ Kr, bf16_t* __restrict__ Vt) {
  int lane = threadIdx.x & 31, wave = threadIdx.x >> 5;
  int rt = blockIdx.x, mat = blockIdx.y, half = lane >> 4;
  const float* W = (mat == 0) ? Wq : (mat == 1) ? Wk : Wv;
  const float scale = (mat == 0) ? 0.0625f : 1.0f;
  for (int t = 0; t < 2; ++t) {
    int ct = wave * 2 + t;
    v8f acc = gemm_f32ab(h + rt * 16 * DD, W + ct * 16 * DD, DD, DD / 32, lane);
    int col = ct * 16 + (lane & 15);
    int head = col & 7, hd = col >> 3;   // reshape(N, HD, H): head is fastest
    int n0 = rt * 16 + 8 * half;         // rows n0 .. n0+7 (within one 32-chunk)
    bf16_t* dst;
    size_t base;
    int rs;
    if (mat == 2) {
      int kp0 = 2 * (n0 & 15) + ((n0 >> 4) & 1);   // interleave permutation
      dst = Vt;
      base = (size_t)head * (HDD * N_N) + (size_t)hd * N_N + (n0 & ~31) + kp0;
      rs = 2;
    } else {
      dst = (mat == 1) ? Kr : Qr;
      base = (size_t)head * (N_N * HDD) + (size_t)n0 * HDD + hd;
      rs = HDD;
    }
#pragma unroll
    for (int r = 0; r < 8; ++r)
      dst[base + (size_t)r * rs] = (bf16_t)(acc[r] * scale);
  }
}

// ---------------------------------------------------------------------------
// Kernel 2: masked attention, streaming over N in 32-col chunks.
// One 16-row block per workgroup; wave == head (8 waves). Probs are packed
// pairwise (cols c, c+16) with v_cvt_pk_bf16_f32 -> 8 ds_store_b32, reloaded
// as the A operand (2 ds_load_b128) under the interleaved K-permutation that
// Vt already carries. No running max needed: |A*score| is O(1) (1/16 scale).
// ---------------------------------------------------------------------------
__global__ __launch_bounds__(256) void k_attn(
    const float* __restrict__ Amask, const bf16_t* __restrict__ Qr,
    const bf16_t* __restrict__ Kr, const bf16_t* __restrict__ Vt,
    float* __restrict__ attnO) {
  __shared__ unsigned int lds_p[NH][16 * 16];   // [row][word c] = (P[c], P[c+16])
  int lane = threadIdx.x & 31, head = threadIdx.x >> 5;
  int half = lane >> 4, col0 = lane & 15;
  int rowbase = blockIdx.x * 16;
  int rbase2 = rowbase + 8 * half;

  const bf16_t* Qh = Qr + head * (N_N * HDD);
  const bf16_t* Kh = Kr + head * (N_N * HDD);
  const bf16_t* Vh = Vt + head * (HDD * N_N);
  unsigned int* pl = lds_p[head];
  const float4* plr = reinterpret_cast<const float4*>(pl + (lane & 15) * 16);

  v16bf qa = load_a_bf16(Qh + rowbase * HDD, HDD, lane);
  v8f o0 = {}, o1 = {};
  float lsum[8];
#pragma unroll
  for (int r = 0; r < 8; ++r) lsum[r] = 0.0f;

  for (int j = 0; j < N_N; j += 32) {
    v16bf kb0 = load_b_rows_bf16(Kh + (size_t)j * HDD, HDD, lane);
    v16bf kb1 = load_b_rows_bf16(Kh + (size_t)(j + 16) * HDD, HDD, lane);
    v8f z = {};
    v8f s0 = wmma_bf16(qa, kb0, z);   // scores cols j..j+15
    v8f s1 = wmma_bf16(qa, kb1, z);   // scores cols j+16..j+31
#pragma unroll
    for (int r = 0; r < 8; ++r) {
      const float* arow = Amask + (size_t)(rbase2 + r) * N_N + j + col0;
      float p0 = __expf(arow[0]  * s0[r]);
      float p1 = __expf(arow[16] * s1[r]);
      lsum[r] += p0 + p1;
      // pack (col c, col c+16) into one dword -> effective K interleave
      union { bf16_t b[2]; unsigned int u; } pk;
      pk.b[0] = (bf16_t)p0;
      pk.b[1] = (bf16_t)p1;
      pl[(r + 8 * half) * 16 + col0] = pk.u;
    }
    asm volatile("s_wait_dscnt 0" ::: "memory");
    // A-layout reload: VGPRs 0-7 = words half*8..+7 (contiguous permuted K)
    v16bf pa;
    float4* pa4 = reinterpret_cast<float4*>(&pa);
    pa4[0] = plr[2 * half];
    pa4[1] = plr[2 * half + 1];
    v16bf vb0 = load_b_rows_bf16(Vh + j, N_N, lane);            // hd 0..15
    v16bf vb1 = load_b_rows_bf16(Vh + 16 * N_N + j, N_N, lane); // hd 16..31
    o0 = wmma_bf16(pa, vb0, o0);
    o1 = wmma_bf16(pa, vb1, o1);
  }
  // one cross-lane row-sum at the end (within 16-lane halves), then normalize
#pragma unroll
  for (int r = 0; r < 8; ++r) {
    float l = lsum[r];
    l += __shfl_xor(l, 1, 32);
    l += __shfl_xor(l, 2, 32);
    l += __shfl_xor(l, 4, 32);
    l += __shfl_xor(l, 8, 32);
    float inv = 1.0f / l;
    int n = rbase2 + r;
    attnO[(size_t)n * DD + col0 * 8 + head]        = o0[r] * inv;  // col = hd*H + head
    attnO[(size_t)n * DD + (col0 + 16) * 8 + head] = o1[r] * inv;
  }
}

// ---------------------------------------------------------------------------
// Kernel 3/6: out = A @ W^T + res, plus per-column sum / sumsq via f32 atomics.
// ASTRIDE (== K == W row length) and operand type are compile-time so the K
// loop fully unrolls.
// ---------------------------------------------------------------------------
template <int ASTRIDE, bool ABF16>
__global__ __launch_bounds__(256) void k_gemm_res_stats(
    const void* __restrict__ Ap, const float* __restrict__ W,
    const float* __restrict__ res, float* __restrict__ out,
    float* __restrict__ s_sum, float* __restrict__ s_sq) {
  int lane = threadIdx.x & 31, wave = threadIdx.x >> 5;
  int rt = blockIdx.x, half = lane >> 4;
  constexpr int KS = ASTRIDE / 32;
  for (int t = 0; t < 2; ++t) {
    int ct = wave * 2 + t;
    v8f acc;
    if constexpr (ABF16)
      acc = gemm_bf16a((const bf16_t*)Ap + (size_t)rt * 16 * ASTRIDE,
                       W + (size_t)ct * 16 * ASTRIDE, ASTRIDE, KS, lane);
    else
      acc = gemm_f32ab((const float*)Ap + (size_t)rt * 16 * ASTRIDE,
                       W + (size_t)ct * 16 * ASTRIDE, ASTRIDE, KS, lane);
    int col = ct * 16 + (lane & 15);
    float cs = 0.0f, sq = 0.0f;
#pragma unroll
    for (int r = 0; r < 8; ++r) {
      int n = rt * 16 + r + 8 * half;
      float v = acc[r] + res[(size_t)n * DD + col];
      out[(size_t)n * DD + col] = v;
      cs += v; sq += v * v;
    }
    cs += __shfl_xor(cs, 16, 32);   // fold the two 8-row halves (same col set)
    sq += __shfl_xor(sq, 16, 32);
    if (lane < 16) {
      atomicAdd(&s_sum[col], cs);
      atomicAdd(&s_sq[col], sq);
    }
  }
}

// ---------------------------------------------------------------------------
// Kernel 4/7: finalize BatchNorm (biased var) and apply; optional bf16 copy.
// ---------------------------------------------------------------------------
__global__ void k_bn_apply(const float* __restrict__ x, const float* __restrict__ ssum,
                           const float* __restrict__ ssq, const float* __restrict__ g,
                           const float* __restrict__ b, float* __restrict__ outf,
                           bf16_t* __restrict__ outb) {
  int idx = blockIdx.x * blockDim.x + threadIdx.x;
  if (idx >= N_N * DD) return;
  int c = idx & (DD - 1);
  float mean = ssum[c] * (1.0f / N_N);
  float var  = ssq[c] * (1.0f / N_N) - mean * mean;
  float rstd = rsqrtf(var + EPSV);
  float v = (x[idx] - mean) * rstd * g[c] + b[c];
  if (outf) outf[idx] = v;
  if (outb) outb[idx] = (bf16_t)v;
}

// ---------------------------------------------------------------------------
// Kernel 5: FFN first layer, relu(hn @ W1^T), bf16 output [N, 512].
// 8 waves x 4 col-tiles = 512 cols.
// ---------------------------------------------------------------------------
__global__ __launch_bounds__(256) void k_ffn1(
    const bf16_t* __restrict__ hnb, const float* __restrict__ W1,
    bf16_t* __restrict__ zb) {
  int lane = threadIdx.x & 31, wave = threadIdx.x >> 5;
  int rt = blockIdx.x, half = lane >> 4;
  for (int t = 0; t < 4; ++t) {
    int ct = wave * 4 + t;
    v8f acc = gemm_bf16a(hnb + (size_t)rt * 16 * DD, W1 + (size_t)ct * 16 * DD,
                         DD, DD / 32, lane);
    int col = ct * 16 + (lane & 15);
#pragma unroll
    for (int r = 0; r < 8; ++r) {
      int n = rt * 16 + r + 8 * half;
      float v = acc[r];
      zb[(size_t)n * (2 * DD) + col] = (bf16_t)(v > 0.0f ? v : 0.0f);
    }
  }
}

// ---------------------------------------------------------------------------
// Launch: inputs (all f32): A, h, Wq, Wk, Wv, Wo, g1, b1, g2, b2, W1, W2
// ---------------------------------------------------------------------------
extern "C" void kernel_launch(void* const* d_in, const int* in_sizes, int n_in,
                              void* d_out, int out_size, void* d_ws, size_t ws_size,
                              hipStream_t stream) {
  const float* A  = (const float*)d_in[0];
  const float* h  = (const float*)d_in[1];
  const float* Wq = (const float*)d_in[2];
  const float* Wk = (const float*)d_in[3];
  const float* Wv = (const float*)d_in[4];
  const float* Wo = (const float*)d_in[5];
  const float* g1 = (const float*)d_in[6];
  const float* b1 = (const float*)d_in[7];
  const float* g2 = (const float*)d_in[8];
  const float* b2 = (const float*)d_in[9];
  const float* W1 = (const float*)d_in[10];
  const float* W2 = (const float*)d_in[11];
  float* out = (float*)d_out;

  char* ws = (char*)d_ws;
  const size_t MB = 1024 * 1024;
  bf16_t* Qr    = (bf16_t*)(ws + 0 * MB);   // [H][N][hd] bf16, 2MB
  bf16_t* Kr    = (bf16_t*)(ws + 2 * MB);   // [H][N][hd] bf16, 2MB
  bf16_t* Vt    = (bf16_t*)(ws + 4 * MB);   // [H][hd][perm(N)] bf16, 2MB
  float*  attnO = (float*) (ws + 6 * MB);   // [N][D] f32, 4MB
  float*  y     = (float*) (ws + 10 * MB);  // [N][D] f32, 4MB
  float*  hn    = (float*) (ws + 14 * MB);  // [N][D] f32, 4MB
  bf16_t* hnb   = (bf16_t*)(ws + 18 * MB);  // [N][D] bf16, 2MB
  bf16_t* zb    = (bf16_t*)(ws + 20 * MB);  // [N][2D] bf16, 4MB
  float*  f     = (float*) (ws + 24 * MB);  // [N][D] f32, 4MB
  float*  st    = (float*) (ws + 28 * MB);  // 4 x 256 BN accumulators
  float *bnsum1 = st, *bnsq1 = st + 256, *bnsum2 = st + 512, *bnsq2 = st + 768;

  k_zero<<<1, 1024, 0, stream>>>(st, 1024);
  k_qkv<<<dim3(N_N / 16, 3), 256, 0, stream>>>(h, Wq, Wk, Wv, Qr, Kr, Vt);
  k_attn<<<N_N / 16, 256, 0, stream>>>(A, Qr, Kr, Vt, attnO);
  k_gemm_res_stats<DD, false><<<N_N / 16, 256, 0, stream>>>(attnO, Wo, h, y,
                                                            bnsum1, bnsq1);
  k_bn_apply<<<(N_N * DD) / 256, 256, 0, stream>>>(y, bnsum1, bnsq1, g1, b1, hn, hnb);
  k_ffn1<<<N_N / 16, 256, 0, stream>>>(hnb, W1, zb);
  k_gemm_res_stats<2 * DD, true><<<N_N / 16, 256, 0, stream>>>(zb, W2, hn, f,
                                                               bnsum2, bnsq2);
  k_bn_apply<<<(N_N * DD) / 256, 256, 0, stream>>>(f, bnsum2, bnsq2, g2, b2, out,
                                                   nullptr);
}
